// LSTM_34815004901875
// MI455X (gfx1250) — matrix operand order repaired and checked by reading, not compiled, not run on hardware
//
#include <hip/hip_runtime.h>

typedef __attribute__((ext_vector_type(16))) _Float16 v16h;
typedef __attribute__((ext_vector_type(8)))  _Float16 v8h;
typedef __attribute__((ext_vector_type(8)))  float    v8f;

#define AS0 184   // f16 row stride, phase A (K=160 padded; 368B = odd multiple of 16B)
#define AS1 264   // f16 row stride, phase B (K=256 padded; 528B = odd multiple of 16B)

// ---- fragment loaders (wave32 WMMA 16-bit layouts, CDNA5 ISA §7.12.2) ----
// A 16x32 fragment: lane(row = l&15) loads K = kb..kb+15 contiguous (same order as ldB), kb = (l>>4)*16
__device__ __forceinline__ v16h ldA(const _Float16* p) {
  v8h a0 = *(const v8h*)p;         // 16B aligned
  v8h a1 = *(const v8h*)(p + 8);
  return __builtin_shufflevector(a0, a1, 0,1,2,3,4,5,6,7,8,9,10,11,12,13,14,15);
}
// B 32x16 fragment: lane(col = l&15) loads K = kb..kb+15 contiguous (same order as ldA), kb = (l>>4)*16
__device__ __forceinline__ v16h ldB(const _Float16* p) {
  v8h b0 = *(const v8h*)p;
  v8h b1 = *(const v8h*)(p + 8);
  return __builtin_shufflevector(b0, b1, 0,1,2,3,4,5,6,7,8,9,10,11,12,13,14,15);
}

__device__ __forceinline__ float fexp(float x) {
  return __builtin_amdgcn_exp2f(x * 1.4426950408889634f);
}
__device__ __forceinline__ float fsig(float x) {
  return __builtin_amdgcn_rcpf(1.0f + fexp(-x));
}
__device__ __forceinline__ float ftanh(float x) {
  float ax = fabsf(x);
  float e  = fexp(-2.0f * ax);
  float t  = (1.0f - e) * __builtin_amdgcn_rcpf(1.0f + e);
  return copysignf(t, x);
}

// =====================  Phase A: LSTM layer 0  =====================
// grid 256 (16 batch rows each), block 256 = 8 wave32
__global__ __launch_bounds__(256) void lstm0_kernel(
    const float* __restrict__ x,     // [4096,64,8]
    const float* __restrict__ Wih,   // [512,8]
    const float* __restrict__ Whh,   // [512,128]
    const float* __restrict__ bih,
    const float* __restrict__ bhh,
    _Float16* __restrict__ ys0)      // [4096,64,128] f16
{
  extern __shared__ _Float16 smem[];
  _Float16* sW = smem;               // [512][AS0]  fused [W_hh0 | W_ih0 | 0]
  _Float16* sA = smem + 512 * AS0;   // [16][AS0]   fused [h0 | x_t | 0]

  const int tid  = threadIdx.x;
  const int lane = tid & 31;
  const int w    = tid >> 5;     // wave id: owns hidden cols [w*16, w*16+16)
  const int lo   = lane & 15;
  const int hi   = lane >> 4;
  const int b0   = blockIdx.x * 16;
  const int cm   = tid >> 4;           // copy-out row (0..15)
  const int c8   = (tid & 15) * 8;     // copy-out col base (0..120)

  // stage fused weights to LDS (f16)
  for (int idx = tid; idx < 512 * 160; idx += 256) {
    int n = idx / 160, k = idx - n * 160;
    float v = (k < 128) ? Whh[n * 128 + k]
            : (k < 136) ? Wih[n * 8 + (k - 128)] : 0.0f;
    sW[n * AS0 + k] = (_Float16)v;
  }
  for (int idx = tid; idx < 16 * AS0; idx += 256) sA[idx] = (_Float16)0.0f;
  __syncthreads();

  // hoist ALL B-matrix fragments into registers (loop-invariant weights):
  // 5 ksteps x 4 gate tiles = 20 x v16h = 160 VGPRs
  v16h Bh[20];
#pragma unroll
  for (int s = 0; s < 5; ++s)
#pragma unroll
    for (int g = 0; g < 4; ++g)
      Bh[s * 4 + g] = ldB(sW + (g * 128 + w * 16 + lo) * AS0 + s * 32 + hi * 16);

  if (tid < 128) {                       // x_0
    int m = tid >> 3, i = tid & 7;
    sA[m * AS0 + 128 + i] = (_Float16)x[((size_t)(b0 + m) * 64 + 0) * 8 + i];
  }
  float bias[4];
#pragma unroll
  for (int g = 0; g < 4; ++g) {
    int col = g * 128 + w * 16 + lo;
    bias[g] = bih[col] + bhh[col];
  }
  float c[8];
#pragma unroll
  for (int v = 0; v < 8; ++v) c[v] = 0.0f;
  __syncthreads();

  for (int t = 0; t < 64; ++t) {
    // prefetch x_{t+1} into a register (latency hidden behind WMMAs)
    int tn = (t < 63) ? (t + 1) : 63;
    float xnext = 0.0f;
    if (tid < 128)
      xnext = x[((size_t)(b0 + (tid >> 3)) * 64 + tn) * 8 + (tid & 7)];

    v8f acc[4];
#pragma unroll
    for (int g = 0; g < 4; ++g)
#pragma unroll
      for (int v = 0; v < 8; ++v) acc[g][v] = bias[g];

    const _Float16* aRow = sA + lo * AS0;
#pragma unroll
    for (int s = 0; s < 5; ++s) {                      // K = 160
      v16h aF = ldA(aRow + s * 32 + hi * 16);
#pragma unroll
      for (int g = 0; g < 4; ++g)
        acc[g] = __builtin_amdgcn_wmma_f32_16x16x32_f16(
            false, aF, false, Bh[s * 4 + g], (short)0, acc[g], false, false);
    }
    __syncthreads();   // all A-buffer reads done

#pragma unroll
    for (int v = 0; v < 8; ++v) {
      float iv = fsig(acc[0][v]);
      float fv = fsig(acc[1][v]);
      float gv = ftanh(acc[2][v]);
      float ov = fsig(acc[3][v]);
      c[v] = fv * c[v] + iv * gv;
      float h = ov * ftanh(c[v]);
      int m = v + hi * 8, n = w * 16 + lo;
      sA[m * AS0 + n] = (_Float16)h;                   // next-step A
    }
    if (t < 63 && tid < 128)                           // store prefetched x_{t+1}
      sA[(tid >> 3) * AS0 + 128 + (tid & 7)] = (_Float16)xnext;
    __syncthreads();   // h & x visible block-wide

    // coalesced ys0 copy-out for step t (16B/thread); overlaps next WMMAs.
    {
      v8h hv = *(const v8h*)(sA + cm * AS0 + c8);
      *(v8h*)(ys0 + ((size_t)(b0 + cm) * 64 + t) * 128 + c8) = hv;
    }
  }
}

// =====================  Phase B: LSTM layer 1 + head  =====================
__global__ __launch_bounds__(256) void lstm1_head_kernel(
    const _Float16* __restrict__ ys0,  // [4096,64,128]
    const float* __restrict__ Wih,     // [512,128]
    const float* __restrict__ Whh,     // [512,128]
    const float* __restrict__ bih,
    const float* __restrict__ bhh,
    const float* __restrict__ Wd1,     // [5,128]
    const float* __restrict__ bd1,     // [5]
    const float* __restrict__ Wd2,     // [1,5]
    const float* __restrict__ bd2,     // [1]
    float* __restrict__ out)           // [4096,1]
{
  extern __shared__ _Float16 smem[];
  _Float16* sW = smem;               // [512][AS1]  fused [W_hh1 | W_ih1]
  _Float16* sA = smem + 512 * AS1;   // [16][AS1]   fused [h1 | y0_t]

  const int tid  = threadIdx.x;
  const int lane = tid & 31;
  const int w    = tid >> 5;
  const int lo   = lane & 15;
  const int hi   = lane >> 4;
  const int b0   = blockIdx.x * 16;
  const int cm   = tid >> 4;           // staging row (0..15)
  const int c8   = (tid & 15) * 8;     // staging col base

  for (int idx = tid; idx < 512 * 256; idx += 256) {
    int n = idx >> 8, k = idx & 255;
    float v = (k < 128) ? Whh[n * 128 + k] : Wih[n * 128 + (k - 128)];
    sW[n * AS1 + k] = (_Float16)v;
  }
  for (int idx = tid; idx < 16 * AS1; idx += 256) sA[idx] = (_Float16)0.0f;
  __syncthreads();

  // hold HALF the B fragments in registers (ksteps 0..3): 16 x v16h = 128 VGPRs
  v16h Bh[16];
#pragma unroll
  for (int s = 0; s < 4; ++s)
#pragma unroll
    for (int g = 0; g < 4; ++g)
      Bh[s * 4 + g] = ldB(sW + (g * 128 + w * 16 + lo) * AS1 + s * 32 + hi * 16);

  {                                          // preload y0_0 (16B per thread)
    *(v8h*)(sA + cm * AS1 + 128 + c8) =
        *(const v8h*)(ys0 + ((size_t)(b0 + cm) * 64 + 0) * 128 + c8);
  }
  float bias[4];
#pragma unroll
  for (int g = 0; g < 4; ++g) {
    int col = g * 128 + w * 16 + lo;
    bias[g] = bih[col] + bhh[col];
  }
  float c[8];
#pragma unroll
  for (int v = 0; v < 8; ++v) c[v] = 0.0f;
  __syncthreads();

  for (int t = 0; t < 64; ++t) {
    // prefetch y0_{t+1}; load issues before the WMMA block (L2 latency hidden)
    int tn = (t < 63) ? (t + 1) : 63;
    v8h ynext = *(const v8h*)(ys0 + ((size_t)(b0 + cm) * 64 + tn) * 128 + c8);

    v8f acc[4];
#pragma unroll
    for (int g = 0; g < 4; ++g)
#pragma unroll
      for (int v = 0; v < 8; ++v) acc[g][v] = bias[g];

    const _Float16* aRow = sA + lo * AS1;
#pragma unroll
    for (int s = 0; s < 8; ++s) {                      // K = 256
      v16h aF = ldA(aRow + s * 32 + hi * 16);
      if (s < 4) {
#pragma unroll
        for (int g = 0; g < 4; ++g)
          acc[g] = __builtin_amdgcn_wmma_f32_16x16x32_f16(
              false, aF, false, Bh[s * 4 + g], (short)0, acc[g], false, false);
      } else {
#pragma unroll
        for (int g = 0; g < 4; ++g) {
          v16h bF = ldB(sW + (g * 128 + w * 16 + lo) * AS1 + s * 32 + hi * 16);
          acc[g] = __builtin_amdgcn_wmma_f32_16x16x32_f16(
              false, aF, false, bF, (short)0, acc[g], false, false);
        }
      }
    }
    __syncthreads();

#pragma unroll
    for (int v = 0; v < 8; ++v) {
      float iv = fsig(acc[0][v]);
      float fv = fsig(acc[1][v]);
      float gv = ftanh(acc[2][v]);
      float ov = fsig(acc[3][v]);
      c[v] = fv * c[v] + iv * gv;
      float h = ov * ftanh(c[v]);
      int m = v + hi * 8, n = w * 16 + lo;
      sA[m * AS1 + n] = (_Float16)h;
    }
    if (t < 63)                                        // stage prefetched y0_{t+1}
      *(v8h*)(sA + cm * AS1 + 128 + c8) = ynext;
    __syncthreads();
  }

  // MLP head: relu(h) -> relu(.@Wd1^T+bd1) -> .@Wd2^T+bd2
  if (tid < 16) {
    int m = tid;
    float s0 = bd1[0], s1 = bd1[1], s2 = bd1[2], s3 = bd1[3], s4 = bd1[4];
    for (int k = 0; k < 128; ++k) {
      float hv = fmaxf((float)sA[m * AS1 + k], 0.0f);
      s0 += hv * Wd1[0 * 128 + k];
      s1 += hv * Wd1[1 * 128 + k];
      s2 += hv * Wd1[2 * 128 + k];
      s3 += hv * Wd1[3 * 128 + k];
      s4 += hv * Wd1[4 * 128 + k];
    }
    float o = bd2[0];
    o += fmaxf(s0, 0.0f) * Wd2[0];
    o += fmaxf(s1, 0.0f) * Wd2[1];
    o += fmaxf(s2, 0.0f) * Wd2[2];
    o += fmaxf(s3, 0.0f) * Wd2[3];
    o += fmaxf(s4, 0.0f) * Wd2[4];
    out[b0 + m] = o;
  }
}

extern "C" void kernel_launch(void* const* d_in, const int* in_sizes, int n_in,
                              void* d_out, int out_size, void* d_ws, size_t ws_size,
                              hipStream_t stream) {
  (void)in_sizes; (void)n_in; (void)out_size; (void)ws_size;
  const float* x    = (const float*)d_in[0];
  const float* Wih0 = (const float*)d_in[1];
  const float* Whh0 = (const float*)d_in[2];
  const float* bih0 = (const float*)d_in[3];
  const float* bhh0 = (const float*)d_in[4];
  const float* Wih1 = (const float*)d_in[5];
  const float* Whh1 = (const float*)d_in[6];
  const float* bih1 = (const float*)d_in[7];
  const float* bhh1 = (const float*)d_in[8];
  const float* Wd1  = (const float*)d_in[9];
  const float* bd1  = (const float*)d_in[10];
  const float* Wd2  = (const float*)d_in[11];
  const float* bd2  = (const float*)d_in[12];

  _Float16* ys0 = (_Float16*)d_ws;   // [4096,64,128] f16 = 64 MB

  size_t ldsA = (size_t)(512 + 16) * AS0 * sizeof(_Float16);  // 194,304 B
  size_t ldsB = (size_t)(512 + 16) * AS1 * sizeof(_Float16);  // 278,784 B

  hipLaunchKernelGGL(lstm0_kernel, dim3(256), dim3(256), ldsA, stream,
                     x, Wih0, Whh0, bih0, bhh0, ys0);
  hipLaunchKernelGGL(lstm1_head_kernel, dim3(256), dim3(256), ldsB, stream,
                     ys0, Wih1, Whh1, bih1, bhh1, Wd1, bd1, Wd2, bd2,
                     (float*)d_out);
}
